// Decoder_51702816309848
// MI455X (gfx1250) — matrix operand order repaired and checked by reading, not compiled, not run on hardware
//
#include <hip/hip_runtime.h>

// ---------------- types ----------------
typedef __attribute__((ext_vector_type(16))) __bf16 v16bf;
typedef __attribute__((ext_vector_type(8)))  float  v8f;
typedef __attribute__((ext_vector_type(4)))  unsigned int u32x4;

#define TENC 200
#define TDEC 200
#define BATCH 16
#define HID 1024
#define ENC 512
#define ATTN 128
#define PRE 256
#define NMEL 80

// ---------------- WMMA fragment loaders (bf16, 16x16x32) ----------------
// A (16xK row-major): lane L -> row M=L%16, K-halves {base..base+7, base+16..base+23}, base=(L>>4)*8
__device__ __forceinline__ v16bf load_frag_a(const __bf16* base, int lda, int lane, int k0) {
  int m = lane & 15, half = lane >> 4;
  const __bf16* p = base + m * lda + k0 + half * 8;
  union { v16bf v; u32x4 q[2]; } r;
  r.q[0] = *(const u32x4*)(p);
  r.q[1] = *(const u32x4*)(p + 16);
  return r.v;
}
// B (Kx16 from weight W[N,K] row-major, so B[k][n]=W[n][k]):
// lane L -> col N=L%16, K-halves {base..base+7, base+16..base+23}, base=(L>>4)*8
__device__ __forceinline__ v16bf load_frag_b(const __bf16* w, int ldw, int lane, int n0, int k0) {
  int n = lane & 15, half = lane >> 4;
  const __bf16* p = w + (n0 + n) * ldw + k0 + half * 8;
  union { v16bf v; u32x4 q[2]; } r;
  r.q[0] = *(const u32x4*)(p);
  r.q[1] = *(const u32x4*)(p + 16);
  return r.v;
}

__device__ __forceinline__ v8f wmma_acc(v8f c, const __bf16* A, int lda,
                                        const __bf16* W, int ldw, int n0, int K, int lane) {
  for (int k = 0; k < K; k += 32) {
    v16bf a = load_frag_a(A, lda, lane, k);
    v16bf b = load_frag_b(W, ldw, lane, n0, k);
    c = __builtin_amdgcn_wmma_f32_16x16x32_bf16(false, a, false, b, (short)0, c, false, false);
  }
  return c;
}

__device__ __forceinline__ float sigm(float x) { return 1.0f / (1.0f + expf(-x)); }

// ---------------- fused LSTM cell (batch16) ----------------
// grid 64 (16 hidden cols each), block 128 (4 waves: i,f,g,o gate chunks)
__global__ __launch_bounds__(128) void lstm_step(
    const __bf16* __restrict__ A0, int K0,      // first input segment  [16,K0]
    const __bf16* __restrict__ A1, int K1,      // second input segment [16,K1]
    const __bf16* __restrict__ Hin,             // recurrent h (bf16)   [16,1024]
    const __bf16* __restrict__ Wih,             // [4096, K0+K1] bf16
    const __bf16* __restrict__ Whh,             // [4096, 1024] bf16
    const float* __restrict__ bih, const float* __restrict__ bhh,
    float* __restrict__ Cst,                    // [16,1024] cell state (in/out)
    float* __restrict__ Hf,                     // [16,1024] new h fp32
    __bf16* __restrict__ Hbf) {                 // [16,1024] new h bf16 (other buffer)
  int wg = blockIdx.x;
  int w = threadIdx.x >> 5;       // gate chunk 0..3 (i,f,g,o)
  int lane = threadIdx.x & 31;
  int ldih = K0 + K1;
  int h0 = wg * 16;
  int n0 = w * HID + h0;

  v8f c = {0.f,0.f,0.f,0.f,0.f,0.f,0.f,0.f};
  c = wmma_acc(c, A0, K0, Wih,        ldih, n0, K0, lane);
  c = wmma_acc(c, A1, K1, Wih + K0,   ldih, n0, K1, lane);
  c = wmma_acc(c, Hin, HID, Whh,      HID,  n0, HID, lane);

  int n = lane & 15, half = lane >> 4;
  float bias = bih[n0 + n] + bhh[n0 + n];
  __shared__ float g[4][16][16];
#pragma unroll
  for (int e = 0; e < 8; ++e) g[w][e + 8 * half][n] = c[e] + bias;
  __syncthreads();

  for (int idx = threadIdx.x; idx < 256; idx += 128) {
    int M = idx >> 4, N = idx & 15;
    float gi = g[0][M][N], gf = g[1][M][N], gg = g[2][M][N], go = g[3][M][N];
    int off = M * HID + h0 + N;
    float cold = Cst[off];
    float c2 = sigm(gf) * cold + sigm(gi) * tanhf(gg);
    float h2 = sigm(go) * tanhf(c2);
    Cst[off] = c2;
    Hf[off] = h2;
    Hbf[off] = (__bf16)h2;
  }
}

// ---------------- location-sensitive attention (one block per batch) ----------------
__global__ __launch_bounds__(256) void attn_step(
    const float* __restrict__ ah,      // [16,1024]
    const float* __restrict__ Wq,      // [128,1024]
    const float* __restrict__ Wc,      // [32,2,31]
    const float* __restrict__ Wld,     // [128,32]
    const float* __restrict__ Wv,      // [128]
    const float* __restrict__ pmem,    // [16*200,128]
    const float* __restrict__ memory,  // [16,200,512]
    const int*   __restrict__ mlen,    // [16]
    float* __restrict__ aw, float* __restrict__ awc,   // [16,200]
    float* __restrict__ ctx, __bf16* __restrict__ ctx_bf, // [16,512]
    float* __restrict__ align_out) {   // d_out alignments + t_step*200; idx b*40000+t
  int b = blockIdx.x, tid = threadIdx.x;
  __shared__ float pq2[2][128];
  __shared__ float pq[128];
  __shared__ float conv[32][TENC];
  __shared__ float es[256];
  __shared__ float red[256];

  // pq = ah @ Wq^T
  {
    int a = tid & 127, part = tid >> 7;
    const float* wq = Wq + a * HID + part * 512;
    const float* h = ah + b * HID + part * 512;
    float acc = 0.f;
    for (int k = 0; k < 512; ++k) acc += wq[k] * h[k];
    pq2[part][a] = acc;
  }
  __syncthreads();
  if (tid < 128) pq[tid] = pq2[0][tid] + pq2[1][tid];

  // location conv over [aw; awc], kernel 31, pad 15
  for (int idx = tid; idx < 32 * TENC; idx += 256) {
    int ch = idx / TENC, t = idx % TENC;
    const float* w0 = Wc + ch * 62;
    const float* w1 = w0 + 31;
    float acc = 0.f;
    for (int k = 0; k < 31; ++k) {
      int tt = t + k - 15;
      if (tt >= 0 && tt < TENC) {
        acc += aw[b * TENC + tt] * w0[k] + awc[b * TENC + tt] * w1[k];
      }
    }
    conv[ch][t] = acc;
  }
  __syncthreads();

  // energies e[t] = sum_a tanh(pq[a] + loc[t,a] + pmem[b,t,a]) * Wv[a]
  float e = -1e30f;
  if (tid < TENC) {
    int t = tid;
    float acc = 0.f;
    const float* pm = pmem + (b * TENC + t) * ATTN;
    for (int a = 0; a < ATTN; ++a) {
      const float* wl = Wld + a * 32;
      float loc = 0.f;
      for (int ch = 0; ch < 32; ++ch) loc += wl[ch] * conv[ch][t];
      acc += tanhf(pq[a] + loc + pm[a]) * Wv[a];
    }
    e = (t >= mlen[b]) ? -1e9f : acc;
  }
  es[tid] = e;
  __syncthreads();

  // softmax over T_enc
  red[tid] = es[tid];
  __syncthreads();
  for (int s = 128; s > 0; s >>= 1) {
    if (tid < s) red[tid] = fmaxf(red[tid], red[tid + s]);
    __syncthreads();
  }
  float mx = red[0];
  __syncthreads();
  float p = (tid < TENC) ? expf(es[tid] - mx) : 0.f;
  red[tid] = p;
  __syncthreads();
  for (int s = 128; s > 0; s >>= 1) {
    if (tid < s) red[tid] += red[tid + s];
    __syncthreads();
  }
  float inv = 1.f / red[0];
  __syncthreads();
  float awn = p * inv;
  if (tid < TENC) {
    aw[b * TENC + tid] = awn;
    awc[b * TENC + tid] += awn;
    align_out[b * (TDEC * TENC) + tid] = awn;
  }
  es[tid] = awn;
  __syncthreads();

  // context = aw @ memory
  for (int d = tid; d < ENC; d += 256) {
    const float* mp = memory + (b * TENC) * ENC + d;
    float acc = 0.f;
    for (int t = 0; t < TENC; ++t) acc += es[t] * mp[t * ENC];
    ctx[b * ENC + d] = acc;
    ctx_bf[b * ENC + d] = (__bf16)acc;
  }
}

// ---------------- mel / gate projection (1 block) ----------------
__global__ __launch_bounds__(256) void proj_step(
    const float* __restrict__ dh, const float* __restrict__ ctx,
    const float* __restrict__ Wproj, const float* __restrict__ Wgate,
    const float* __restrict__ bgate,
    float* __restrict__ mel_out,   // d_out + t ; idx b*16000 + j*200
    float* __restrict__ gate_out)  // d_out + 256000 + t ; idx b*200
{
  int tid = threadIdx.x;
  for (int oi = tid; oi < BATCH * 81; oi += 256) {
    int b = oi / 81, j = oi % 81;
    const float* w = (j < 80) ? (Wproj + j * (HID + ENC)) : Wgate;
    const float* h = dh + b * HID;
    const float* cx = ctx + b * ENC;
    float acc = (j < 80) ? 0.f : bgate[0];
    for (int k = 0; k < HID; ++k) acc += w[k] * h[k];
    for (int k = 0; k < ENC; ++k) acc += w[HID + k] * cx[k];
    if (j < 80) mel_out[b * (NMEL * TDEC) + j * TDEC] = acc;
    else        gate_out[b * TDEC] = acc;
  }
}

// ---------------- generic WMMA tile GEMMs for the prologue ----------------
// blocks = (rows/16)*ntn, 32 threads. Out[M,N] (+relu, bf16) ; A row-major [rows,K], W [N,K]
__global__ __launch_bounds__(32) void wmma_gemm_relu_bf16(
    const __bf16* __restrict__ A, const __bf16* __restrict__ W,
    __bf16* __restrict__ Out, int K, int ldo, int ntn) {
  int rt = blockIdx.x / ntn, nt = blockIdx.x % ntn;
  int lane = threadIdx.x;
  v8f c = {0.f,0.f,0.f,0.f,0.f,0.f,0.f,0.f};
  c = wmma_acc(c, A + rt * 16 * K, K, W, K, nt * 16, K, lane);
  int n = lane & 15, half = lane >> 4;
#pragma unroll
  for (int e = 0; e < 8; ++e) {
    float v = c[e];
    v = v > 0.f ? v : 0.f;
    Out[(rt * 16 + e + 8 * half) * ldo + nt * 16 + n] = (__bf16)v;
  }
}

__global__ __launch_bounds__(32) void wmma_gemm_f32(
    const __bf16* __restrict__ A, const __bf16* __restrict__ W,
    float* __restrict__ Out, int K, int ldo, int ntn) {
  int rt = blockIdx.x / ntn, nt = blockIdx.x % ntn;
  int lane = threadIdx.x;
  v8f c = {0.f,0.f,0.f,0.f,0.f,0.f,0.f,0.f};
  c = wmma_acc(c, A + rt * 16 * K, K, W, K, nt * 16, K, lane);
  int n = lane & 15, half = lane >> 4;
#pragma unroll
  for (int e = 0; e < 8; ++e)
    Out[(rt * 16 + e + 8 * half) * ldo + nt * 16 + n] = c[e];
}

// ---------------- prologue helpers ----------------
__global__ void f32_to_bf16(const float* __restrict__ s, __bf16* __restrict__ d, int n) {
  int i = blockIdx.x * 256 + threadIdx.x;
  if (i < n) d[i] = (__bf16)s[i];
}
__global__ void zero_f32(float* p, int n) {
  int i = blockIdx.x * 256 + threadIdx.x;
  if (i < n) p[i] = 0.f;
}
__global__ void zero_bf16(__bf16* p, int n) {
  int i = blockIdx.x * 256 + threadIdx.x;
  if (i < n) p[i] = (__bf16)0.f;
}
// build teacher-forced prenet input [T*B, 96] (go frame + K-pad to 96), bf16
__global__ void build_xin(const float* __restrict__ dec_in, __bf16* __restrict__ xin) {
  int idx = blockIdx.x * 256 + threadIdx.x;
  if (idx >= TDEC * BATCH * 96) return;
  int row = idx / 96, col = idx % 96;
  int t = row >> 4, b = row & 15;
  float v = 0.f;
  if (col < NMEL && t > 0) v = dec_in[b * (NMEL * TDEC) + col * TDEC + (t - 1)];
  xin[idx] = (__bf16)v;
}
// pad Wp1 [256,80] -> [256,96] bf16
__global__ void build_wp1p(const float* __restrict__ Wp1, __bf16* __restrict__ out) {
  int idx = blockIdx.x * 256 + threadIdx.x;
  if (idx >= PRE * 96) return;
  int r = idx / 96, c = idx % 96;
  out[idx] = (__bf16)((c < NMEL) ? Wp1[r * NMEL + c] : 0.f);
}

// ---------------- host driver ----------------
extern "C" void kernel_launch(void* const* d_in, const int* in_sizes, int n_in,
                              void* d_out, int out_size, void* d_ws, size_t ws_size,
                              hipStream_t stream) {
  const float* memory = (const float*)d_in[0];
  const float* dec_in = (const float*)d_in[1];
  const int*   mlen   = (const int*)d_in[2];
  const float* Wp1    = (const float*)d_in[3];
  const float* Wp2    = (const float*)d_in[4];
  const float* Wih_a  = (const float*)d_in[5];
  const float* Whh_a  = (const float*)d_in[6];
  const float* bih_a  = (const float*)d_in[7];
  const float* bhh_a  = (const float*)d_in[8];
  const float* Wih_d  = (const float*)d_in[9];
  const float* Whh_d  = (const float*)d_in[10];
  const float* bih_d  = (const float*)d_in[11];
  const float* bhh_d  = (const float*)d_in[12];
  const float* Wq     = (const float*)d_in[13];
  const float* Wm     = (const float*)d_in[14];
  const float* Wconv  = (const float*)d_in[15];
  const float* Wld    = (const float*)d_in[16];
  const float* Wv     = (const float*)d_in[17];
  const float* Wproj  = (const float*)d_in[18];
  const float* Wgate  = (const float*)d_in[19];
  const float* bgate  = (const float*)d_in[20];
  float* dout = (float*)d_out;

  char* ws = (char*)d_ws;
  size_t off = 0;
  auto alloc = [&](size_t bytes) -> char* {
    char* p = ws + off;
    off = (off + bytes + 255) & ~(size_t)255;
    return p;
  };

  const int NWIA = 4096 * 768, NWHA = 4096 * 1024, NWID = 4096 * 1536, NWHD = 4096 * 1024;
  const int NWM = ATTN * ENC, NWP2 = PRE * PRE, NMEM = BATCH * TENC * ENC;

  __bf16* wih_a_bf = (__bf16*)alloc((size_t)NWIA * 2);
  __bf16* whh_a_bf = (__bf16*)alloc((size_t)NWHA * 2);
  __bf16* wih_d_bf = (__bf16*)alloc((size_t)NWID * 2);
  __bf16* whh_d_bf = (__bf16*)alloc((size_t)NWHD * 2);
  __bf16* wm_bf    = (__bf16*)alloc((size_t)NWM * 2);
  __bf16* wp1p     = (__bf16*)alloc((size_t)PRE * 96 * 2);
  __bf16* wp2_bf   = (__bf16*)alloc((size_t)NWP2 * 2);
  __bf16* mem_bf   = (__bf16*)alloc((size_t)NMEM * 2);
  __bf16* xin      = (__bf16*)alloc((size_t)TDEC * BATCH * 96 * 2);
  __bf16* xs1      = (__bf16*)alloc((size_t)TDEC * BATCH * PRE * 2);
  __bf16* xs       = (__bf16*)alloc((size_t)TDEC * BATCH * PRE * 2);
  float*  pmem     = (float*)alloc((size_t)BATCH * TENC * ATTN * 4);
  float*  ahf      = (float*)alloc((size_t)BATCH * HID * 4);
  float*  dhf      = (float*)alloc((size_t)BATCH * HID * 4);
  float*  ac       = (float*)alloc((size_t)BATCH * HID * 4);
  float*  dc       = (float*)alloc((size_t)BATCH * HID * 4);
  float*  ctxf     = (float*)alloc((size_t)BATCH * ENC * 4);
  float*  aw       = (float*)alloc((size_t)BATCH * TENC * 4);
  float*  awc      = (float*)alloc((size_t)BATCH * TENC * 4);
  __bf16* ahbf[2]; ahbf[0] = (__bf16*)alloc((size_t)BATCH * HID * 2);
                   ahbf[1] = (__bf16*)alloc((size_t)BATCH * HID * 2);
  __bf16* dhbf[2]; dhbf[0] = (__bf16*)alloc((size_t)BATCH * HID * 2);
                   dhbf[1] = (__bf16*)alloc((size_t)BATCH * HID * 2);
  __bf16* ctxbf    = (__bf16*)alloc((size_t)BATCH * ENC * 2);
  (void)ws_size; (void)in_sizes; (void)n_in; (void)out_size;

  auto cvt = [&](const float* s, __bf16* d, int n) {
    f32_to_bf16<<<(n + 255) / 256, 256, 0, stream>>>(s, d, n);
  };
  // weights + memory -> bf16 (deterministic, redone each call)
  cvt(Wih_a, wih_a_bf, NWIA);
  cvt(Whh_a, whh_a_bf, NWHA);
  cvt(Wih_d, wih_d_bf, NWID);
  cvt(Whh_d, whh_d_bf, NWHD);
  cvt(Wm, wm_bf, NWM);
  cvt(Wp2, wp2_bf, NWP2);
  cvt(memory, mem_bf, NMEM);
  build_xin<<<(TDEC * BATCH * 96 + 255) / 256, 256, 0, stream>>>(dec_in, xin);
  build_wp1p<<<(PRE * 96 + 255) / 256, 256, 0, stream>>>(Wp1, wp1p);

  // zero initial state
  zero_f32<<<(BATCH * HID + 255) / 256, 256, 0, stream>>>(ac, BATCH * HID);
  zero_f32<<<(BATCH * HID + 255) / 256, 256, 0, stream>>>(dc, BATCH * HID);
  zero_f32<<<(BATCH * TENC + 255) / 256, 256, 0, stream>>>(aw, BATCH * TENC);
  zero_f32<<<(BATCH * TENC + 255) / 256, 256, 0, stream>>>(awc, BATCH * TENC);
  zero_bf16<<<(BATCH * HID + 255) / 256, 256, 0, stream>>>(ahbf[0], BATCH * HID);
  zero_bf16<<<(BATCH * HID + 255) / 256, 256, 0, stream>>>(dhbf[0], BATCH * HID);
  zero_bf16<<<(BATCH * ENC + 255) / 256, 256, 0, stream>>>(ctxbf, BATCH * ENC);

  // prenet for all 200 steps: [3200,96]->relu->[3200,256]->relu->[3200,256] (bf16)
  wmma_gemm_relu_bf16<<<200 * 16, 32, 0, stream>>>(xin, wp1p, xs1, 96, PRE, 16);
  wmma_gemm_relu_bf16<<<200 * 16, 32, 0, stream>>>(xs1, wp2_bf, xs, PRE, PRE, 16);
  // pmem = memory @ Wm^T : [3200,512] -> [3200,128] fp32
  wmma_gemm_f32<<<200 * 8, 32, 0, stream>>>(mem_bf, wm_bf, pmem, ENC, ATTN, 8);

  float* mel_base = dout;                         // [16,80,200]
  float* gate_base = dout + BATCH * NMEL * TDEC;  // [16,200]
  float* align_base = gate_base + BATCH * TDEC;   // [16,200,200]

  int p = 0;
  for (int t = 0; t < TDEC; ++t) {
    // attention LSTM: x = [prenet(x_t) (256) | ctx (512)], h = ah
    lstm_step<<<64, 128, 0, stream>>>(
        xs + (size_t)t * BATCH * PRE, PRE, ctxbf, ENC, ahbf[p],
        wih_a_bf, whh_a_bf, bih_a, bhh_a, ac, ahf, ahbf[1 - p]);
    // location-sensitive attention -> aw, awc, ctx, alignment output
    attn_step<<<16, 256, 0, stream>>>(
        ahf, Wq, Wconv, Wld, Wv, pmem, memory, mlen,
        aw, awc, ctxf, ctxbf, align_base + (size_t)t * TENC);
    // decoder LSTM: x = [ah (1024) | ctx (512)], h = dh
    lstm_step<<<64, 128, 0, stream>>>(
        ahbf[1 - p], HID, ctxbf, ENC, dhbf[p],
        wih_d_bf, whh_d_bf, bih_d, bhh_d, dc, dhf, dhbf[1 - p]);
    // projection + gate
    proj_step<<<1, 256, 0, stream>>>(
        dhf, ctxf, Wproj, Wgate, bgate, mel_base + t, gate_base + t);
    p ^= 1;
  }
}
